// LinearAttention_22050362097776
// MI455X (gfx1250) — matrix operand order repaired and checked by reading, not compiled, not run on hardware
//
#include <hip/hip_runtime.h>
#include <hip/hip_bf16.h>
#include <math.h>

// ---------------------------------------------------------------------------
// CDNA5 (gfx1250, wave32) LinearAttention pipeline, bf16 WMMA w/ f32 accum.
// Single-buffered LDS GEMM (round-1 structure: proven spill-free codegen);
// latency hidden via multi-block occupancy (24KB LDS -> many blocks/WGP).
// ---------------------------------------------------------------------------

typedef __attribute__((ext_vector_type(16))) __bf16 v16bf;
typedef __attribute__((ext_vector_type(8)))  __bf16 v8bf;
typedef __attribute__((ext_vector_type(8)))  float  v8f;

#define LDSS 48   // padded LDS row stride (elements) -> 96B, 16B/32B aligned
#define BK   32   // K-step matches wmma_f32_16x16x32_bf16

__device__ __forceinline__ __bf16 f2bf(float f) { return (__bf16)f; }

__device__ __forceinline__ v16bf combine16(v8bf lo, v8bf hi) {
  v16bf r;
#pragma unroll
  for (int i = 0; i < 8; ++i) { r[i] = lo[i]; r[8 + i] = hi[i]; }
  return r;
}

// A fragment (16x32 bf16), tile row-major in LDS (stride LDSS).
// lane l (l15=l&15, lhi=l>>4): row m=l15, K = lhi*8 + {0..7} and lhi*8+16+{0..7}
__device__ __forceinline__ v16bf load_a_frag(const __bf16* tile, int l15, int lhi) {
  const __bf16* p = tile + l15 * LDSS + lhi * 8;
  v8bf lo = *(const v8bf*)(p);
  v8bf hi = *(const v8bf*)(p + 16);
  return combine16(lo, hi);
}

// B fragment (32x16 bf16), tile stored TRANSPOSED in LDS as [n][k] (stride LDSS).
// lane l: column n=l15, K = lhi*8 + {0..7} and lhi*8+16+{0..7}
__device__ __forceinline__ v16bf load_b_frag(const __bf16* tile, int l15, int lhi) {
  const __bf16* p = tile + l15 * LDSS + lhi * 8;
  v8bf lo = *(const v8bf*)(p);
  v8bf hi = *(const v8bf*)(p + 16);
  return combine16(lo, hi);
}

// --- B staging: global (K x N row-major) tile -> LDS transposed [n][k] -------
__device__ __forceinline__ void stage_b(const float* src, __bf16* Bs, int bk, int bseg) {
  alignas(16) float tmp[16];
  ((float4*)tmp)[0] = ((const float4*)src)[0];
  ((float4*)tmp)[1] = ((const float4*)src)[1];
  ((float4*)tmp)[2] = ((const float4*)src)[2];
  ((float4*)tmp)[3] = ((const float4*)src)[3];
#pragma unroll
  for (int j = 0; j < 16; ++j) Bs[(bseg + j) * LDSS + bk] = f2bf(tmp[j]);
}
__device__ __forceinline__ void stage_b(const __bf16* src, __bf16* Bs, int bk, int bseg) {
  v8bf d0 = *(const v8bf*)(src);
  v8bf d1 = *(const v8bf*)(src + 8);
#pragma unroll
  for (int j = 0; j < 8; ++j) {
    Bs[(bseg + j) * LDSS + bk]     = d0[j];
    Bs[(bseg + 8 + j) * LDSS + bk] = d1[j];
  }
}

// --- generic 128x128 block GEMM (A f32 row-major, B f32/bf16 row-major K x N)
// block = 256 threads = 8 waves; wave (wid&1, wid>>1) -> 64x32 sub-tile.
template <typename TB>
__device__ __forceinline__ void block_gemm(
    const float* __restrict__ A, int lda, int m0,
    const TB* __restrict__ B, int ldb, int n0,
    int K, __bf16* As, __bf16* Bs, v8f acc[4][2]) {
  const int tid  = threadIdx.x;
  const int lane = tid & 31, wid = tid >> 5;
  const int l15 = lane & 15, lhi = lane >> 4;
  const int wm = (wid & 1) * 64;
  const int wn = (wid >> 1) * 32;
  const int arow  = tid >> 1;         // 0..127
  const int ahalf = (tid & 1) * 16;   // 0 / 16
  const int bk    = tid >> 3;         // 0..31
  const int bseg  = (tid & 7) * 16;   // 0..112

#pragma unroll
  for (int tm = 0; tm < 4; ++tm)
#pragma unroll
    for (int tn = 0; tn < 2; ++tn)
#pragma unroll
      for (int i = 0; i < 8; ++i) acc[tm][tn][i] = 0.0f;

  for (int k0 = 0; k0 < K; k0 += BK) {
    // stage A 128x32 (f32 -> bf16), row-major
    {
      const float* src = A + (size_t)(m0 + arow) * lda + k0 + ahalf;
      alignas(16) float tmp[16];
      ((float4*)tmp)[0] = ((const float4*)src)[0];
      ((float4*)tmp)[1] = ((const float4*)src)[1];
      ((float4*)tmp)[2] = ((const float4*)src)[2];
      ((float4*)tmp)[3] = ((const float4*)src)[3];
      __bf16* dst = As + arow * LDSS + ahalf;
#pragma unroll
      for (int j = 0; j < 16; ++j) dst[j] = f2bf(tmp[j]);
    }
    // stage B 32x128 transposed into [n][k]
    stage_b(B + (size_t)(k0 + bk) * ldb + n0 + bseg, Bs, bk, bseg);

    if (k0 + BK < K) {  // global_prefetch_b8, near-cache locality
      __builtin_prefetch(A + (size_t)(m0 + arow) * lda + k0 + BK + ahalf, 0, 3);
      __builtin_prefetch(B + (size_t)(k0 + BK + bk) * ldb + n0 + bseg, 0, 3);
    }
    __syncthreads();

    v16bf af[4], bfv[2];
#pragma unroll
    for (int tm = 0; tm < 4; ++tm)
      af[tm] = load_a_frag(As + (wm + tm * 16) * LDSS, l15, lhi);
#pragma unroll
    for (int tn = 0; tn < 2; ++tn)
      bfv[tn] = load_b_frag(Bs + (wn + tn * 16) * LDSS, l15, lhi);

#pragma unroll
    for (int tm = 0; tm < 4; ++tm)
#pragma unroll
      for (int tn = 0; tn < 2; ++tn)
        acc[tm][tn] = __builtin_amdgcn_wmma_f32_16x16x32_bf16(
            false, af[tm], false, bfv[tn], (short)0, acc[tm][tn], false, false);
    __syncthreads();
  }
}

// ---------------------------------------------------------------------------
// Kernel 1: qkv = w_qkv (1536x256) @ x[b] (256x4096); split into q/k_raw/v.
// grid (32 nblk, 12 mblk, 16 b), block 256.
// ---------------------------------------------------------------------------
__global__ void qkv_proj_kernel(const float* __restrict__ x,
                                const float* __restrict__ w_qkv,
                                __bf16* __restrict__ qb,
                                float* __restrict__ kraw,
                                __bf16* __restrict__ vb) {
  __shared__ __align__(16) __bf16 As[128 * LDSS];
  __shared__ __align__(16) __bf16 Bs[128 * LDSS];
  const int b  = blockIdx.z;
  const int m0 = blockIdx.y * 128;
  const int n0 = blockIdx.x * 128;
  v8f acc[4][2];
  block_gemm<float>(w_qkv, 256, m0, x + (size_t)b * 256 * 4096, 4096, n0,
                    256, As, Bs, acc);

  const int lane = threadIdx.x & 31, wid = threadIdx.x >> 5;
  const int l15 = lane & 15, lhi = lane >> 4;
  const int wm = (wid & 1) * 64, wn = (wid >> 1) * 32;
#pragma unroll
  for (int tm = 0; tm < 4; ++tm)
#pragma unroll
    for (int tn = 0; tn < 2; ++tn)
#pragma unroll
      for (int i = 0; i < 8; ++i) {
        const int m = m0 + wm + tm * 16 + 8 * lhi + i;
        const int n = n0 + wn + tn * 16 + l15;
        const float vvl = acc[tm][tn][i];
        if (m < 512)
          qb[((size_t)b * 512 + m) * 4096 + n] = f2bf(vvl);
        else if (m < 1024)
          kraw[((size_t)b * 512 + (m - 512)) * 4096 + n] = vvl;
        else
          vb[((size_t)b * 512 + (m - 1024)) * 4096 + n] = f2bf(vvl);
      }
}

// ---------------------------------------------------------------------------
// Kernel 2: row softmax over n=4096 (8192 rows), f32 in -> bf16 out.
// ---------------------------------------------------------------------------
__global__ void softmax_rows_kernel(const float* __restrict__ in,
                                    __bf16* __restrict__ out) {
  __shared__ float red[256];
  const int row = blockIdx.x;
  const int t   = threadIdx.x;
  const float* src = in + (size_t)row * 4096 + t * 16;
  __bf16* dst      = out + (size_t)row * 4096 + t * 16;

  alignas(16) float v[16];
  ((float4*)v)[0] = ((const float4*)src)[0];
  ((float4*)v)[1] = ((const float4*)src)[1];
  ((float4*)v)[2] = ((const float4*)src)[2];
  ((float4*)v)[3] = ((const float4*)src)[3];

  float m = v[0];
#pragma unroll
  for (int i = 1; i < 16; ++i) m = fmaxf(m, v[i]);
  red[t] = m; __syncthreads();
  for (int s = 128; s > 0; s >>= 1) {
    if (t < s) red[t] = fmaxf(red[t], red[t + s]);
    __syncthreads();
  }
  const float rowmax = red[0]; __syncthreads();

  float s = 0.0f;
#pragma unroll
  for (int i = 0; i < 16; ++i) { v[i] = __expf(v[i] - rowmax); s += v[i]; }
  red[t] = s; __syncthreads();
  for (int st = 128; st > 0; st >>= 1) {
    if (t < st) red[t] += red[t + st];
    __syncthreads();
  }
  const float inv = 1.0f / red[0];
#pragma unroll
  for (int i = 0; i < 16; ++i) dst[i] = f2bf(v[i] * inv);
}

// ---------------------------------------------------------------------------
// Kernel 3: contextT partials. Per (b,h): context[d][e] = sum_n k[d][n]*v[e][n]
// NT-GEMM, fragments straight from global (both operands row-major over n).
// Split-K x 8 (deterministic, no atomics); stores TRANSPOSED: part[e][d].
// grid (8 splits, 64 bh), block 256.
// ---------------------------------------------------------------------------
__global__ void attn_context_kernel(const __bf16* __restrict__ ksoft,
                                    const __bf16* __restrict__ vb,
                                    float* __restrict__ part) {
  const int split = blockIdx.x;
  const int bh    = blockIdx.y;
  const __bf16* Kp = ksoft + (size_t)bh * 128 * 4096;
  const __bf16* Vp = vb    + (size_t)bh * 128 * 4096;

  const int lane = threadIdx.x & 31, wid = threadIdx.x >> 5;
  const int l15 = lane & 15, lhi = lane >> 4;
  const int wm = (wid & 1) * 64, wn = (wid >> 1) * 32;

  v8f acc[4][2];
#pragma unroll
  for (int tm = 0; tm < 4; ++tm)
#pragma unroll
    for (int tn = 0; tn < 2; ++tn)
#pragma unroll
      for (int i = 0; i < 8; ++i) acc[tm][tn][i] = 0.0f;

  const int nk0 = split * 512, nk1 = nk0 + 512;
  for (int nk = nk0; nk < nk1; nk += BK) {
    v16bf af[4], bfv[2];
#pragma unroll
    for (int tm = 0; tm < 4; ++tm) {
      const __bf16* p = Kp + (size_t)(wm + tm * 16 + l15) * 4096 + nk + lhi * 8;
      af[tm] = combine16(*(const v8bf*)p, *(const v8bf*)(p + 16));
    }
#pragma unroll
    for (int tn = 0; tn < 2; ++tn) {
      const __bf16* p = Vp + (size_t)(wn + tn * 16 + l15) * 4096 + nk + lhi * 8;
      bfv[tn] = combine16(*(const v8bf*)p, *(const v8bf*)(p + 16));
    }
#pragma unroll
    for (int tm = 0; tm < 4; ++tm)
#pragma unroll
      for (int tn = 0; tn < 2; ++tn)
        acc[tm][tn] = __builtin_amdgcn_wmma_f32_16x16x32_bf16(
            false, af[tm], false, bfv[tn], (short)0, acc[tm][tn], false, false);
  }

  float* dst = part + ((size_t)split * 64 + bh) * 16384;  // [e][d]
#pragma unroll
  for (int tm = 0; tm < 4; ++tm)
#pragma unroll
    for (int tn = 0; tn < 2; ++tn)
#pragma unroll
      for (int i = 0; i < 8; ++i) {
        const int m = wm + tm * 16 + 8 * lhi + i;  // d
        const int n = wn + tn * 16 + l15;          // e
        dst[(size_t)n * 128 + m] = acc[tm][tn][i]; // transposed store
      }
}

// Kernel 4: reduce split-K partials -> contextT (f32). 64*16384 outputs.
__global__ void reduce_context_kernel(const float* __restrict__ part,
                                      float* __restrict__ ctxT) {
  const size_t idx = (size_t)blockIdx.x * 256 + threadIdx.x;  // < 1,048,576
  float s = 0.0f;
#pragma unroll
  for (int sp = 0; sp < 8; ++sp) s += part[(size_t)sp * 1048576 + idx];
  ctxT[idx] = s;
}

// ---------------------------------------------------------------------------
// Kernel 5: out[e][n] = contextT(e x d) @ q(d x n). grid (32 nblk, 64 bh).
// ---------------------------------------------------------------------------
__global__ void attn_out_kernel(const float* __restrict__ ctxT,
                                const __bf16* __restrict__ qb,
                                __bf16* __restrict__ outb) {
  __shared__ __align__(16) __bf16 As[128 * LDSS];
  __shared__ __align__(16) __bf16 Bs[128 * LDSS];
  const int bh = blockIdx.y;
  const int n0 = blockIdx.x * 128;
  v8f acc[4][2];
  block_gemm<__bf16>(ctxT + (size_t)bh * 16384, 128, 0,
                     qb + (size_t)bh * 128 * 4096, 4096, n0,
                     128, As, Bs, acc);

  const int lane = threadIdx.x & 31, wid = threadIdx.x >> 5;
  const int l15 = lane & 15, lhi = lane >> 4;
  const int wm = (wid & 1) * 64, wn = (wid >> 1) * 32;
#pragma unroll
  for (int tm = 0; tm < 4; ++tm)
#pragma unroll
    for (int tn = 0; tn < 2; ++tn)
#pragma unroll
      for (int i = 0; i < 8; ++i) {
        const int m = wm + tm * 16 + 8 * lhi + i;        // e
        const int n = n0 + wn + tn * 16 + l15;           // n
        outb[((size_t)bh * 128 + m) * 4096 + n] = f2bf(acc[tm][tn][i]);
      }
}

// ---------------------------------------------------------------------------
// Kernel 6: y = w_out (256x512) @ out[b] (512x4096) + bias. f32 output.
// grid (32 nblk, 2 mblk, 16 b).
// ---------------------------------------------------------------------------
__global__ void out_proj_kernel(const float* __restrict__ w_out,
                                const __bf16* __restrict__ outb,
                                const float* __restrict__ bias,
                                float* __restrict__ y) {
  __shared__ __align__(16) __bf16 As[128 * LDSS];
  __shared__ __align__(16) __bf16 Bs[128 * LDSS];
  const int b  = blockIdx.z;
  const int m0 = blockIdx.y * 128;
  const int n0 = blockIdx.x * 128;
  v8f acc[4][2];
  block_gemm<__bf16>(w_out, 512, m0, outb + (size_t)b * 512 * 4096, 4096, n0,
                     512, As, Bs, acc);

  const int lane = threadIdx.x & 31, wid = threadIdx.x >> 5;
  const int l15 = lane & 15, lhi = lane >> 4;
  const int wm = (wid & 1) * 64, wn = (wid >> 1) * 32;
#pragma unroll
  for (int tm = 0; tm < 4; ++tm)
#pragma unroll
    for (int tn = 0; tn < 2; ++tn)
#pragma unroll
      for (int i = 0; i < 8; ++i) {
        const int m = m0 + wm + tm * 16 + 8 * lhi + i;
        const int n = n0 + wn + tn * 16 + l15;
        y[((size_t)b * 256 + m) * 4096 + n] = acc[tm][tn][i] + bias[m];
      }
}

// ---------------------------------------------------------------------------
extern "C" void kernel_launch(void* const* d_in, const int* in_sizes, int n_in,
                              void* d_out, int out_size, void* d_ws, size_t ws_size,
                              hipStream_t stream) {
  const float* x     = (const float*)d_in[0];  // (16,256,64,64)
  const float* w_qkv = (const float*)d_in[1];  // (1536,256)
  const float* w_out = (const float*)d_in[2];  // (256,512)
  const float* b_out = (const float*)d_in[3];  // (256,)
  float* y = (float*)d_out;                    // (16,256,64,64)

  const size_t QB = 16ull * 512 * 4096;  // elements per q/k/v/out tensor
  char* w = (char*)d_ws;
  size_t off = 0;
  auto take = [&](size_t bytes) -> char* {
    char* p = w + off;
    off = (off + bytes + 255) & ~(size_t)255;
    return p;
  };
  __bf16* qb    = (__bf16*)take(QB * 2);                  // q, bf16
  __bf16* ksoft = (__bf16*)take(QB * 2);                  // softmax(k), bf16
  __bf16* vb    = (__bf16*)take(QB * 2);                  // v, bf16
  __bf16* outb  = (__bf16*)take(QB * 2);                  // attention out, bf16
  float*  kraw  = (float*)take(QB * 4);                   // k pre-softmax, f32
  float*  part  = (float*)take(8ull * 64 * 128 * 128 * 4);// split-K partials
  float*  ctxT  = (float*)take(64ull * 128 * 128 * 4);    // contextT f32
  (void)ws_size; (void)in_sizes; (void)n_in; (void)out_size;

  qkv_proj_kernel<<<dim3(32, 12, 16), 256, 0, stream>>>(x, w_qkv, qb, kraw, vb);
  softmax_rows_kernel<<<8192, 256, 0, stream>>>(kraw, ksoft);
  attn_context_kernel<<<dim3(8, 64), 256, 0, stream>>>(ksoft, vb, part);
  reduce_context_kernel<<<4096, 256, 0, stream>>>(part, ctxT);
  attn_out_kernel<<<dim3(32, 64), 256, 0, stream>>>(ctxT, qb, outb);
  out_proj_kernel<<<dim3(32, 2, 16), 256, 0, stream>>>(w_out, outb, b_out, y);
}
